// SparseAxialCausalAttention_70635032150532
// MI455X (gfx1250) — matrix operand order repaired and checked by reading, not compiled, not run on hardware
//
#include <hip/hip_runtime.h>

typedef __bf16 bf16_t;
typedef __attribute__((ext_vector_type(16))) __bf16 v16bf;
typedef __attribute__((ext_vector_type(8)))  __bf16 v8bf;
typedef __attribute__((ext_vector_type(8)))  float  v8f;
typedef __attribute__((ext_vector_type(4)))  int    v4i;

#define NH    16
#define DH    64
#define IMG   32
#define SEQ   1279
#define NPAD  1280
#define DIM   1024
#define TEXT  256
#define LIMG  288            // 256 text + 32 axial image keys
#define QSCALE 0.125f        // 64^-0.5
#define MASKV (-3.402823466e38f)

union ABfrag { v16bf v; v8bf h[2]; };

static __device__ __forceinline__ v8f wmma_bf16(v16bf a, v16bf b, v8f c) {
    return __builtin_amdgcn_wmma_f32_16x16x32_bf16(false, a, false, b, (short)0, c, false, false);
}

// ---- CDNA5 async global->LDS copy (ASYNCcnt path), guarded so compile can't break ----
#if defined(__has_builtin)
#  if __has_builtin(__builtin_amdgcn_global_load_async_to_lds_b128) && \
      __has_builtin(__builtin_amdgcn_s_wait_asynccnt)
#    define HAS_ASYNC 1
#  endif
#endif
#ifndef HAS_ASYNC
#  define HAS_ASYNC 0
#endif

#if HAS_ASYNC
typedef __attribute__((address_space(1))) v4i gv4i_t;   // global src (b128)
typedef __attribute__((address_space(3))) v4i lv4i_t;   // LDS dst (b128)
#endif

static __device__ __forceinline__ void async_copy_b128(const bf16_t* g, bf16_t* l) {
#if HAS_ASYNC
    gv4i_t* gp = (gv4i_t*)(unsigned long long)g;
    lv4i_t* lp = (lv4i_t*)(unsigned int)(unsigned long long)l;
    __builtin_amdgcn_global_load_async_to_lds_b128(gp, lp, 0, 0);
#else
    *(v8bf*)l = *(const v8bf*)g;   // fallback: load + ds_store, covered by barriers
#endif
}

template <int N>
static __device__ __forceinline__ void wait_async() {
#if HAS_ASYNC
    __builtin_amdgcn_s_wait_asynccnt(N);   // immediate required
#endif
}

// ---------------- conversion kernels (HBM-streaming) ----------------
__global__ void cvt_x_kernel(const float* __restrict__ x, bf16_t* __restrict__ xp) {
    long i = (long)blockIdx.x * 256 + threadIdx.x;
    if (i >= (long)16 * NPAD * DIM) return;
    int  c  = (int)(i % DIM);
    long rb = i / DIM;
    int  r  = (int)(rb % NPAD);
    int  b  = (int)(rb / NPAD);
    float v = (r < SEQ) ? x[((long)b * SEQ + r) * DIM + c] : 0.0f;
    xp[i] = (bf16_t)v;
}

__global__ void cvt_w_kernel(const float* __restrict__ w, bf16_t* __restrict__ o, long n) {
    long i = (long)blockIdx.x * 256 + threadIdx.x;
    if (i < n) o[i] = (bf16_t)w[i];
}

// ---------------- QKV projection: xp[20480x1024] @ w_qkv^T[1024x3072] ----------------
// 4-wave workgroup computes a 128x64 tile. The 64x32 weight k-slice is staged into LDS
// (async b128, double-buffered, ASYNCcnt) once per WG and reused by all 4 waves.
// Each wave: 32x64 accumulator = 8 WMMAs per k-step.
// Epilogue scatters into q[bh][n][64] (scaled), k[bh][n][64], vT[bh][64][n] (transposed).
__global__ __launch_bounds__(128) void gemm_qkv_kernel(
        const bf16_t* __restrict__ xp, const bf16_t* __restrict__ wq,
        bf16_t* __restrict__ qb, bf16_t* __restrict__ kb, bf16_t* __restrict__ vT) {
    __shared__ __align__(64) bf16_t sB[2][64][32];
    const int tid = threadIdx.x, wave = tid >> 5, lane = tid & 31;
    const int m = lane & 15, half = lane >> 4;
    const int mt  = blockIdx.x / 48;     // 160 tiles of 128 rows over M = 16*1280
    const int nt4 = blockIdx.x % 48;     // 48 col-tiles of 64 over N = 3072
    const long rowbase = (long)mt * 128 + wave * 32;
    const bf16_t* arow0 = xp + (rowbase + m) * DIM;
    const bf16_t* arow1 = xp + (rowbase + 16 + m) * DIM;
    const bf16_t* wbase = wq + (long)(nt4 * 64) * DIM;

    // stage one 64x32 B k-slice: 256 16B chunks, 2 per lane
    auto stage = [&](int buf, int kbase) {
#pragma unroll
        for (int i = 0; i < 2; ++i) {
            const int c   = wave * 64 + i * 32 + lane;
            const int row = c >> 2, ch = c & 3;
            async_copy_b128(wbase + (long)row * DIM + kbase + ch * 8, &sB[buf][row][ch * 8]);
        }
    };

    v8f acc[2][4] = {};
    stage(0, 0);
    stage(1, 32);
    for (int ks = 0; ks < DIM / 32; ++ks) {
        if (ks < DIM / 32 - 1) wait_async<2>();  // own 2 instrs for buf[ks&1] done
        else                   wait_async<0>();
        __syncthreads();                          // all waves' slices visible
        const int kbase = ks * 32;
        ABfrag a0, a1;
        a0.h[0] = *(const v8bf*)(arow0 + kbase + 16 * half);
        a0.h[1] = *(const v8bf*)(arow0 + kbase + 8 + 16 * half);
        a1.h[0] = *(const v8bf*)(arow1 + kbase + 16 * half);
        a1.h[1] = *(const v8bf*)(arow1 + kbase + 8 + 16 * half);
#pragma unroll
        for (int sub = 0; sub < 4; ++sub) {
            v16bf b = *(const v16bf*)(&sB[ks & 1][sub * 16 + m][16 * half]);
            acc[0][sub] = wmma_bf16(a0.v, b, acc[0][sub]);
            acc[1][sub] = wmma_bf16(a1.v, b, acc[1][sub]);
        }
        __syncthreads();                          // everyone done reading buf[ks&1]
        if (ks + 2 < DIM / 32) stage(ks & 1, kbase + 64);
    }

#pragma unroll
    for (int mi = 0; mi < 2; ++mi)
#pragma unroll
        for (int sub = 0; sub < 4; ++sub) {
            const int colg = nt4 * 64 + sub * 16 + m;       // D col = lane&15
            const int sec = colg >> 10, cc = colg & 1023;
            const int h = cc >> 6, d = cc & 63;
#pragma unroll
            for (int e = 0; e < 8; ++e) {
                const long r = rowbase + mi * 16 + e + 8 * half;  // D row = e + 8*half
                const int bi = (int)(r / NPAD), n = (int)(r - (long)bi * NPAD);
                const long bh = (long)bi * NH + h;
                const float v = acc[mi][sub][e];
                if (sec == 0)      qb[(bh * NPAD + n) * DH + d] = (bf16_t)(v * QSCALE);
                else if (sec == 1) kb[(bh * NPAD + n) * DH + d] = (bf16_t)v;
                else               vT[(bh * DH + d) * NPAD + n] = (bf16_t)v;
            }
        }
}

// ---------------- attention: one wave per (head, 16-query tile) ----------------
// tiles 0..15  : text queries (causal over text keys)
// tiles 16..79 : image queries, 2 tiles per image row x; keys = 256 text + 32 axial (causal)
__global__ __launch_bounds__(32) void attn_kernel(
        const bf16_t* __restrict__ qb, const bf16_t* __restrict__ kb,
        const bf16_t* __restrict__ vT, bf16_t* __restrict__ y) {
    __shared__ float  sS[16][LIMG];
    __shared__ __align__(32) bf16_t sP[16][LIMG];
    const int lane = threadIdx.x, m = lane & 15, half = lane >> 4;
    const int bh = blockIdx.x / 80, t = blockIdx.x % 80;
    const int bi = bh >> 4, hh = bh & 15;
    const bool is_text = (t < 16);
    int qbase, L, xrow = 0, subq = 0;
    if (is_text) { qbase = t * 16; L = qbase + 16; }
    else { const int ti = t - 16; xrow = ti >> 1; subq = ti & 1;
           qbase = TEXT + 32 * xrow + 16 * subq; L = LIMG; }

    // Q fragment (A operand), resident for the whole tile
    const bf16_t* qrow = qb + ((long)bh * NPAD + qbase + m) * DH;
    ABfrag aq0, aq1;
    aq0.h[0] = *(const v8bf*)(qrow +      16 * half);
    aq0.h[1] = *(const v8bf*)(qrow +  8 + 16 * half);
    aq1.h[0] = *(const v8bf*)(qrow + 32 + 16 * half);
    aq1.h[1] = *(const v8bf*)(qrow + 40 + 16 * half);

    // S = Q K^T, masked, into LDS
    const int nkt = L / 16;
    for (int kt = 0; kt < nkt; ++kt) {
        const int gkb = (kt < 16) ? kt * 16 : TEXT + 32 * xrow + 16 * (kt - 16);
        const bf16_t* krow = kb + ((long)bh * NPAD + gkb + m) * DH + 16 * half;
        v16bf b0 = *(const v16bf*)(krow);
        v16bf b1 = *(const v16bf*)(krow + 32);
        v8f c = {};
        c = wmma_bf16(aq0.v, b0, c);
        c = wmma_bf16(aq1.v, b1, c);
#pragma unroll
        for (int e = 0; e < 8; ++e) {
            const int row = e + 8 * half;
            bool masked;
            if (is_text) masked = (gkb + m) > (qbase + row);
            else         masked = (kt >= 16) && ((16 * (kt - 16) + m) > (16 * subq + row));
            sS[row][kt * 16 + m] = masked ? MASKV : c[e];
        }
    }
    __syncthreads();

    // exact softmax per query row, re-quantize to bf16, zero-pad to 32-key multiple
    const int Lpad = ((L + 31) >> 5) << 5;
    if (lane < 16) {
        const int row = lane;
        float mx = MASKV;
        for (int c = 0; c < L; ++c) mx = fmaxf(mx, sS[row][c]);
        float s = 0.0f;
        for (int c = 0; c < L; ++c) { float e2 = __expf(sS[row][c] - mx); s += e2; sS[row][c] = e2; }
        const float inv = 1.0f / s;
        for (int c = 0; c < L; ++c)    sP[row][c] = (bf16_t)(sS[row][c] * inv);
        for (int c = L; c < Lpad; ++c) sP[row][c] = (bf16_t)0.0f;
    }
    __syncthreads();

    // O = P V : A from LDS, B from transposed V in global
    const int nk32 = Lpad >> 5;
#pragma unroll
    for (int nt = 0; nt < 4; ++nt) {
        const int d = nt * 16 + m;                       // output dim = B col
        const bf16_t* vrow = vT + ((long)bh * DH + d) * NPAD;
        v8f acc = {};
        for (int k32 = 0; k32 < nk32; ++k32) {
            const int lkb = k32 * 32;
            const int gkb = (is_text || lkb < TEXT) ? lkb : TEXT + 32 * xrow;
            ABfrag a;
            a.h[0] = *(const v8bf*)(&sP[m][lkb +     16 * half]);
            a.h[1] = *(const v8bf*)(&sP[m][lkb + 8 + 16 * half]);
            v16bf b = *(const v16bf*)(vrow + gkb + 16 * half);
            acc = wmma_bf16(a.v, b, acc);
        }
#pragma unroll
        for (int e = 0; e < 8; ++e) {
            const int row = e + 8 * half;
            const int qg = qbase + row;
            y[((long)bi * NPAD + qg) * DIM + hh * DH + d] = (bf16_t)acc[e];
        }
    }
}

// ---------------- output projection: y[20480x1024] @ w_out^T + b_out -> fp32 ----------------
// Same 4-wave LDS-staged scheme as gemm_qkv_kernel.
__global__ __launch_bounds__(128) void gemm_out_kernel(
        const bf16_t* __restrict__ y, const bf16_t* __restrict__ wo,
        const float* __restrict__ bias, float* __restrict__ out) {
    __shared__ __align__(64) bf16_t sB[2][64][32];
    const int tid = threadIdx.x, wave = tid >> 5, lane = tid & 31;
    const int m = lane & 15, half = lane >> 4;
    const int mt  = blockIdx.x / 16;     // 160 tiles of 128 rows
    const int nt4 = blockIdx.x % 16;     // 16 col-tiles of 64 over N = 1024
    const long rowbase = (long)mt * 128 + wave * 32;
    const bf16_t* arow0 = y + (rowbase + m) * DIM;
    const bf16_t* arow1 = y + (rowbase + 16 + m) * DIM;
    const bf16_t* wbase = wo + (long)(nt4 * 64) * DIM;

    auto stage = [&](int buf, int kbase) {
#pragma unroll
        for (int i = 0; i < 2; ++i) {
            const int c   = wave * 64 + i * 32 + lane;
            const int row = c >> 2, ch = c & 3;
            async_copy_b128(wbase + (long)row * DIM + kbase + ch * 8, &sB[buf][row][ch * 8]);
        }
    };

    v8f acc[2][4] = {};
    stage(0, 0);
    stage(1, 32);
    for (int ks = 0; ks < DIM / 32; ++ks) {
        if (ks < DIM / 32 - 1) wait_async<2>();
        else                   wait_async<0>();
        __syncthreads();
        const int kbase = ks * 32;
        ABfrag a0, a1;
        a0.h[0] = *(const v8bf*)(arow0 + kbase + 16 * half);
        a0.h[1] = *(const v8bf*)(arow0 + kbase + 8 + 16 * half);
        a1.h[0] = *(const v8bf*)(arow1 + kbase + 16 * half);
        a1.h[1] = *(const v8bf*)(arow1 + kbase + 8 + 16 * half);
#pragma unroll
        for (int sub = 0; sub < 4; ++sub) {
            v16bf b = *(const v16bf*)(&sB[ks & 1][sub * 16 + m][16 * half]);
            acc[0][sub] = wmma_bf16(a0.v, b, acc[0][sub]);
            acc[1][sub] = wmma_bf16(a1.v, b, acc[1][sub]);
        }
        __syncthreads();
        if (ks + 2 < DIM / 32) stage(ks & 1, kbase + 64);
    }

#pragma unroll
    for (int mi = 0; mi < 2; ++mi)
#pragma unroll
        for (int sub = 0; sub < 4; ++sub) {
            const int colg = nt4 * 64 + sub * 16 + m;
            const float bv = bias[colg];
#pragma unroll
            for (int e = 0; e < 8; ++e) {
                const long r = rowbase + mi * 16 + e + 8 * half;
                const int bi = (int)(r / NPAD), n = (int)(r - (long)bi * NPAD);
                if (n < SEQ) out[((long)bi * SEQ + n) * DIM + colg] = acc[mi][sub][e] + bv;
            }
        }
}

// ---------------- host-side launch ----------------
extern "C" void kernel_launch(void* const* d_in, const int* in_sizes, int n_in,
                              void* d_out, int out_size, void* d_ws, size_t ws_size,
                              hipStream_t stream) {
    const float* x     = (const float*)d_in[0];
    const float* w_qkv = (const float*)d_in[1];
    const float* w_out = (const float*)d_in[2];
    const float* b_out = (const float*)d_in[3];
    float* out = (float*)d_out;
    (void)in_sizes; (void)n_in; (void)out_size; (void)ws_size;

    char* ws = (char*)d_ws;
    const size_t SZ_XP  = (size_t)16 * NPAD * DIM * 2;   // 41.9 MB
    const size_t SZ_WQ  = (size_t)3 * DIM * DIM * 2;     //  6.3 MB
    const size_t SZ_WO  = (size_t)DIM * DIM * 2;         //  2.1 MB
    const size_t SZ_HED = (size_t)256 * NPAD * DH * 2;   // 41.9 MB each
    bf16_t* xp  = (bf16_t*)(ws);
    bf16_t* wqb = (bf16_t*)(ws + SZ_XP);
    bf16_t* wob = (bf16_t*)(ws + SZ_XP + SZ_WQ);
    bf16_t* qb  = (bf16_t*)(ws + SZ_XP + SZ_WQ + SZ_WO);
    bf16_t* kb  = (bf16_t*)(ws + SZ_XP + SZ_WQ + SZ_WO + SZ_HED);
    bf16_t* vT  = (bf16_t*)(ws + SZ_XP + SZ_WQ + SZ_WO + 2 * SZ_HED);
    bf16_t* yb  = (bf16_t*)(ws + SZ_XP + SZ_WQ + SZ_WO + 3 * SZ_HED);

    // 1) fp32 -> bf16 (x is zero-padded to 1280 rows)
    cvt_x_kernel<<<(16L * NPAD * DIM + 255) / 256, 256, 0, stream>>>(x, xp);
    cvt_w_kernel<<<((long)3 * DIM * DIM + 255) / 256, 256, 0, stream>>>(w_qkv, wqb, (long)3 * DIM * DIM);
    cvt_w_kernel<<<((long)DIM * DIM + 255) / 256, 256, 0, stream>>>(w_out, wob, (long)DIM * DIM);

    // 2) QKV projection + head split (+ scale folded into q, v transposed)
    gemm_qkv_kernel<<<160 * 48, 128, 0, stream>>>(xp, wqb, qb, kb, vT);

    // 3) sparse axial causal attention (256 heads x 80 query tiles)
    attn_kernel<<<256 * 80, 32, 0, stream>>>(qb, kb, vT, yb);

    // 4) output projection + bias, store fp32 for n < 1279
    gemm_out_kernel<<<160 * 16, 128, 0, stream>>>(yb, wob, b_out, out);
}
